// Net_4518305595713
// MI455X (gfx1250) — matrix operand order repaired and checked by reading, not compiled, not run on hardware
//
#include <hip/hip_runtime.h>
#include <math.h>

// ---------------------------------------------------------------------------
// GIN (5 layers, eps=0) + global_add_pool + MLP head, MI455X (gfx1250, wave32)
//
// - GEMMs: V_WMMA_F32_16X16X32_F16, fully unrolled (templated K), each wave
//   owns NT 16x16 column tiles so one converted A fragment feeds NT WMMAs.
//   Weights pre-converted to f16 transposed [Nout][K] so B fragments are
//   contiguous 32B per lane (same contiguous k order as A).
// - Neighbor aggregation: CSR built ONCE per launch (edge_index is launch-
//   invariant), then 5 atomic-free gather-sum passes (wave per node, b128
//   row loads, all L2-resident: x is 25.6MB << 192MB L2). This replaces
//   5 x 102M f32 atomics with 1.6M int atomics + pure load bandwidth.
// ---------------------------------------------------------------------------

typedef __attribute__((ext_vector_type(16))) _Float16 v16h;
typedef __attribute__((ext_vector_type(8)))  float    v8f;

#define NNODES  50000
#define HN      128
#define H2N     256
#define NGRAPH  256
#define NCLS    10
#define NLAYERS 5
#define BN_EPS  1e-5f

// ---------------- weight convert: f32 [K][Nout] -> f16 [Nout][K] ------------
__global__ void conv_transpose_f16(const float* __restrict__ W,
                                   _Float16* __restrict__ Wt,
                                   int K, int Nout) {
  int t = blockIdx.x * blockDim.x + threadIdx.x;
  if (t >= K * Nout) return;
  int n = t / K;
  int k = t - n * K;
  Wt[t] = (_Float16)W[(size_t)k * Nout + n];
}

// ---------------- CSR build (once per launch) -------------------------------
__global__ void degree_count(const int* __restrict__ dst,
                             int* __restrict__ deg, int E) {
  int e = blockIdx.x * blockDim.x + threadIdx.x;
  if (e >= E) return;
  atomicAdd(&deg[dst[e]], 1);
}

// Single-block chunked exclusive scan: rowptr[0..n] from deg[0..n-1].
__global__ __launch_bounds__(1024)
void exclusive_scan(const int* __restrict__ deg, int* __restrict__ rowptr,
                    int n) {
  __shared__ int buf[1024];
  __shared__ int carry;
  if (threadIdx.x == 0) carry = 0;
  __syncthreads();
  for (int base = 0; base < n; base += 1024) {
    int i = base + threadIdx.x;
    int v = (i < n) ? deg[i] : 0;
    buf[threadIdx.x] = v;
    __syncthreads();
    // Hillis-Steele inclusive scan
    for (int off = 1; off < 1024; off <<= 1) {
      int t = (threadIdx.x >= off) ? buf[threadIdx.x - off] : 0;
      __syncthreads();
      buf[threadIdx.x] += t;
      __syncthreads();
    }
    int c = carry;
    if (i < n) rowptr[i] = c + (buf[threadIdx.x] - v);   // exclusive
    __syncthreads();
    if (threadIdx.x == 0) carry = c + buf[1023];
    __syncthreads();
  }
  if (threadIdx.x == 0) rowptr[n] = carry;
}

__global__ void csr_fill(const int* __restrict__ src,
                         const int* __restrict__ dst,
                         int* __restrict__ cursor,
                         int* __restrict__ srcidx, int E) {
  int e = blockIdx.x * blockDim.x + threadIdx.x;
  if (e >= E) return;
  int pos = atomicAdd(&cursor[dst[e]], 1);
  srcidx[pos] = src[e];
}

// ---------------- GIN aggregation: hsum[n] = x[n] + sum_{j->n} x[j] ---------
// One wave per node; lane owns 4 features; per-wave-uniform edge loop.
__global__ __launch_bounds__(256)
void gather_sum(const float* __restrict__ x,
                const int* __restrict__ rowptr,
                const int* __restrict__ srcidx,
                float* __restrict__ hsum, int N) {
  int node = blockIdx.x * (blockDim.x >> 5) + (threadIdx.x >> 5);
  if (node >= N) return;
  int lane = threadIdx.x & 31;
  int f = lane * 4;
  float4 acc = *(const float4*)(x + (size_t)node * HN + f);  // + x[n] (eps=0)
  int beg = rowptr[node];
  int end = rowptr[node + 1];
  for (int j = beg; j < end; ++j) {
    int s = srcidx[j];                                       // wave-uniform
    const float4 v = *(const float4*)(x + (size_t)s * HN + f);
    acc.x += v.x; acc.y += v.y; acc.z += v.z; acc.w += v.w;
  }
  *(float4*)(hsum + (size_t)node * HN + f) = acc;
}

// ---------------- global_add_pool: p[batch[n]] += x[n] ----------------------
__global__ void pool_add(const float* __restrict__ x,
                         const int* __restrict__ batch,
                         float* __restrict__ p, int N) {
  long long t = (long long)blockIdx.x * blockDim.x + threadIdx.x;
  int node = (int)(t >> 5);
  if (node >= N) return;
  int f = (int)(t & 31) * 4;
  int g = batch[node];
  const float4 val = *(const float4*)(x + (size_t)node * HN + f);
  float* o = p + (size_t)g * HN + f;
  atomicAdd(o + 0, val.x);
  atomicAdd(o + 1, val.y);
  atomicAdd(o + 2, val.z);
  atomicAdd(o + 3, val.w);
}

// ---------------- fused GEMM + bias + eval-BN + ReLU (WMMA f16) -------------
// out[M,Nout] = relu(BN(A @ W + bias)), Wt = f16 [Nout][K].
// Wave handles 16 rows x (NT*16) cols; block = 4 waves; fully unrolled K.
// grid = (M/16, Nout/(64*NT)), blockDim = 128.
template <int K, int NT>
__global__ __launch_bounds__(128)
void gemm_bn_relu_wmma(const float* __restrict__ A,
                       const _Float16* __restrict__ Wt,
                       const float* __restrict__ bias,
                       const float* __restrict__ bn_g,
                       const float* __restrict__ bn_b,
                       const float* __restrict__ bn_m,
                       const float* __restrict__ bn_v,
                       float* __restrict__ out,
                       int Nout) {
  const int lane  = threadIdx.x & 31;
  const int wave  = threadIdx.x >> 5;
  const int tileM = blockIdx.x;
  const int colBase = (blockIdx.y * 4 + wave) * (16 * NT);

  const int lcol  = lane & 15;
  const int khalf = lane >> 4;                  // 0 or 1 (K-half select)
  const int mrow  = tileM * 16 + lcol;          // A row this lane supplies

  const float*    arow  = A + (size_t)mrow * K;
  const _Float16* brow0 = Wt + (size_t)(colBase + lcol) * K;

  v8f acc[NT] = {};

#pragma unroll
  for (int kk = 0; kk < K / 32; ++kk) {
    // A fragment (16x32, this code's contiguous k order, same as B): lane holds K [kb..kb+15]
    // contiguous, kb = 32*kk + 16*khalf. f32 b128 loads -> f16 in-reg.
    const int kbA = kk * 32 + khalf * 16;
    float4 p0 = *(const float4*)(arow + kbA);
    float4 p1 = *(const float4*)(arow + kbA + 4);
    float4 q0 = *(const float4*)(arow + kbA + 8);
    float4 q1 = *(const float4*)(arow + kbA + 12);
    v16h a;
    a[0]  = (_Float16)p0.x; a[1]  = (_Float16)p0.y;
    a[2]  = (_Float16)p0.z; a[3]  = (_Float16)p0.w;
    a[4]  = (_Float16)p1.x; a[5]  = (_Float16)p1.y;
    a[6]  = (_Float16)p1.z; a[7]  = (_Float16)p1.w;
    a[8]  = (_Float16)q0.x; a[9]  = (_Float16)q0.y;
    a[10] = (_Float16)q0.z; a[11] = (_Float16)q0.w;
    a[12] = (_Float16)q1.x; a[13] = (_Float16)q1.y;
    a[14] = (_Float16)q1.z; a[15] = (_Float16)q1.w;

    // B fragment: lane = column, K [kbB..kbB+15] contiguous (transposed f16).
    const int kbB = kk * 32 + khalf * 16;
#pragma unroll
    for (int t = 0; t < NT; ++t) {
      v16h b = *(const v16h*)(brow0 + (size_t)t * 16 * K + kbB);
      acc[t] = __builtin_amdgcn_wmma_f32_16x16x32_f16(
          /*neg_a=*/false, a, /*neg_b=*/false, b,
          /*c_mod=*/(short)0, acc[t], /*reuse_a=*/false, /*reuse_b=*/false);
    }
  }

  // Epilogue: y = relu(acc*scale + add), folding bias + BN per column.
  const int row0 = tileM * 16 + khalf * 8;      // C/D layout: VGPR r -> row
#pragma unroll
  for (int t = 0; t < NT; ++t) {
    const int ncol = colBase + t * 16 + lcol;
    const float scale = bn_g[ncol] * rsqrtf(bn_v[ncol] + BN_EPS);
    const float add   = (bias[ncol] - bn_m[ncol]) * scale + bn_b[ncol];
#pragma unroll
    for (int r = 0; r < 8; ++r) {
      float y = acc[t][r] * scale + add;
      y = y > 0.0f ? y : 0.0f;
      out[(size_t)(row0 + r) * Nout + ncol] = y;
    }
  }
}

// ---------------- final 128->10 linear + log_softmax ------------------------
__global__ void head_final(const float* __restrict__ h,
                           const float* __restrict__ Wl2,
                           const float* __restrict__ bl2,
                           float* __restrict__ out) {
  int g = blockIdx.x;
  int lane = threadIdx.x;
  float logit = -INFINITY;
  if (lane < NCLS) {
    float s = bl2[lane];
    for (int k = 0; k < HN; ++k)
      s += h[(size_t)g * HN + k] * Wl2[k * NCLS + lane];
    logit = s;
  }
  float mx = logit;
  for (int off = 16; off >= 1; off >>= 1)
    mx = fmaxf(mx, __shfl_xor(mx, off, 32));
  float ex = (lane < NCLS) ? __expf(logit - mx) : 0.0f;
  float sum = ex;
  for (int off = 16; off >= 1; off >>= 1)
    sum += __shfl_xor(sum, off, 32);
  if (lane < NCLS)
    out[(size_t)g * NCLS + lane] = (logit - mx) - __logf(sum);
}

// ---------------------------------------------------------------------------
extern "C" void kernel_launch(void* const* d_in, const int* in_sizes, int n_in,
                              void* d_out, int out_size, void* d_ws, size_t ws_size,
                              hipStream_t stream) {
  (void)n_in; (void)out_size; (void)ws_size;

  const float* x     = (const float*)d_in[0];
  const int*   ei    = (const int*)d_in[1];
  const int*   batch = (const int*)d_in[2];
  const float* W1  = (const float*)d_in[3];
  const float* b1  = (const float*)d_in[4];
  const float* g1  = (const float*)d_in[5];
  const float* be1 = (const float*)d_in[6];
  const float* m1  = (const float*)d_in[7];
  const float* v1  = (const float*)d_in[8];
  const float* W2  = (const float*)d_in[9];
  const float* b2  = (const float*)d_in[10];
  const float* gO  = (const float*)d_in[11];
  const float* bO  = (const float*)d_in[12];
  const float* mO  = (const float*)d_in[13];
  const float* vO  = (const float*)d_in[14];
  const float* Wl1 = (const float*)d_in[15];
  const float* bl1 = (const float*)d_in[16];
  const float* gb1 = (const float*)d_in[17];
  const float* bb1 = (const float*)d_in[18];
  const float* mb1 = (const float*)d_in[19];
  const float* vb1 = (const float*)d_in[20];
  const float* Wl2 = (const float*)d_in[21];
  const float* bl2 = (const float*)d_in[22];

  const int E = in_sizes[1] / 2;
  const int* src = ei;
  const int* dst = ei + E;

  // ---- workspace carve-up (256B aligned) ----
  char* ws = (char*)d_ws;
  size_t o = 0;
  auto take = [&](size_t bytes) -> char* {
    char* p = ws + o;
    o += (bytes + 255) & ~(size_t)255;
    return p;
  };
  float*    xbuf   = (float*)take((size_t)NNODES * HN  * sizeof(float));
  float*    h2     = (float*)take((size_t)NNODES * H2N * sizeof(float));
  float*    hsum   = (float*)take((size_t)NNODES * HN  * sizeof(float));
  float*    pbuf   = (float*)take((size_t)NGRAPH * HN  * sizeof(float));
  float*    hh     = (float*)take((size_t)NGRAPH * HN  * sizeof(float));
  _Float16* W1t    = (_Float16*)take((size_t)NLAYERS * HN * H2N * sizeof(_Float16));
  _Float16* W2t    = (_Float16*)take((size_t)NLAYERS * H2N * HN * sizeof(_Float16));
  _Float16* Wl1t   = (_Float16*)take((size_t)HN * HN * sizeof(_Float16));
  int*      deg    = (int*)take((size_t)NNODES * sizeof(int));
  int*      rowptr = (int*)take(((size_t)NNODES + 1) * sizeof(int));
  int*      cursor = (int*)take((size_t)NNODES * sizeof(int));
  int*      srcidx = (int*)take((size_t)E * sizeof(int));

  // ---- convert weights to f16 transposed [Nout][K] ----
  const int WN = HN * H2N;                       // 32768 per layer (both dirs)
  for (int l = 0; l < NLAYERS; ++l) {
    conv_transpose_f16<<<(WN + 255) / 256, 256, 0, stream>>>(
        W1 + (size_t)l * WN, W1t + (size_t)l * WN, HN, H2N);
    conv_transpose_f16<<<(WN + 255) / 256, 256, 0, stream>>>(
        W2 + (size_t)l * WN, W2t + (size_t)l * WN, H2N, HN);
  }
  conv_transpose_f16<<<(HN * HN + 255) / 256, 256, 0, stream>>>(Wl1, Wl1t, HN, HN);

  // ---- build CSR once (edge_index is launch-invariant) ----
  hipMemsetAsync(deg, 0, (size_t)NNODES * sizeof(int), stream);
  degree_count<<<(E + 255) / 256, 256, 0, stream>>>(dst, deg, E);
  exclusive_scan<<<1, 1024, 0, stream>>>(deg, rowptr, NNODES);
  hipMemcpyAsync(cursor, rowptr, (size_t)NNODES * sizeof(int),
                 hipMemcpyDeviceToDevice, stream);
  csr_fill<<<(E + 255) / 256, 256, 0, stream>>>(src, dst, cursor, srcidx, E);

  // ---- GIN layers ----
  const int gatherBlocks = (NNODES + 7) / 8;     // 8 waves (nodes) per block
  const float* xin = x;
  for (int l = 0; l < NLAYERS; ++l) {
    // hsum = x + sum_{j->i} x_j  (atomic-free gather)
    gather_sum<<<gatherBlocks, 256, 0, stream>>>(xin, rowptr, srcidx,
                                                 hsum, NNODES);

    // GEMM1: [N,128] @ [128,256], NT=4 -> block covers 256 cols
    dim3 grid1(NNODES / 16, 1);
    gemm_bn_relu_wmma<HN, 4><<<grid1, 128, 0, stream>>>(
        hsum, W1t + (size_t)l * WN,
        b1 + (size_t)l * H2N, g1 + (size_t)l * H2N, be1 + (size_t)l * H2N,
        m1 + (size_t)l * H2N, v1 + (size_t)l * H2N,
        h2, H2N);

    // GEMM2: [N,256] @ [256,128], NT=2 -> block covers 128 cols
    dim3 grid2(NNODES / 16, 1);
    gemm_bn_relu_wmma<H2N, 2><<<grid2, 128, 0, stream>>>(
        h2, W2t + (size_t)l * WN,
        b2 + (size_t)l * HN, gO + (size_t)l * HN, bO + (size_t)l * HN,
        mO + (size_t)l * HN, vO + (size_t)l * HN,
        xbuf, HN);

    xin = xbuf;
  }

  // ---- readout ----
  hipMemsetAsync(pbuf, 0, (size_t)NGRAPH * HN * sizeof(float), stream);
  long long tN = (long long)NNODES * 32;
  pool_add<<<(int)((tN + 255) / 256), 256, 0, stream>>>(xin, batch, pbuf, NNODES);

  dim3 gridh(NGRAPH / 16, 1);
  gemm_bn_relu_wmma<HN, 2><<<gridh, 128, 0, stream>>>(
      pbuf, Wl1t, bl1, gb1, bb1, mb1, vb1,
      hh, NGRAPH /* Nout (row stride) == NGRAPH == 256 */);

  head_final<<<NGRAPH, 32, 0, stream>>>(hh, Wl2, bl2, (float*)d_out);
}
